// lstm_model_74715251081471
// MI455X (gfx1250) — matrix operand order repaired and checked by reading, not compiled, not run on hardware
//
#include <hip/hip_runtime.h>
#include <hip/hip_bf16.h>

// ---------------------------------------------------------------------------
// Stacked-LSTM (3 x 512) inference for gfx1250 using v_wmma_f32_16x16x32_f16.
//   phase 0: subsample + dense(8->128) + tanh            (scalar, tiny)
//   phase 1: pack all W/U weights into WMMA B-fragment layout (f16)
//   per layer: big GEMM xW+b over all 51200 tokens (WMMA), then 200 sequential
//              fused steps: z = xw_t + h@U (WMMA), gates, c/h update.
//   heads: two 512->60 dots (scalar, tiny)
// ---------------------------------------------------------------------------

typedef __attribute__((ext_vector_type(16))) _Float16 v16h;
typedef __attribute__((ext_vector_type(8)))  _Float16 v8h;
typedef __attribute__((ext_vector_type(8)))  float    v8f;

#define B_SZ   256
#define T_SEQ  200
#define DMODEL 512
#define NGATE  2048          // 4*DMODEL
#define NT     128           // NGATE/16 column tiles
#define M_TOK  (B_SZ*T_SEQ)  // 51200

__device__ __forceinline__ v8f wmma_f16(v16h a, v16h b, v8f c) {
  return __builtin_amdgcn_wmma_f32_16x16x32_f16(
      /*neg_a=*/false, a, /*neg_b=*/false, b,
      /*c_mod=*/(short)0, c, /*reuse_a=*/false, /*reuse_b=*/false);
}

// Load one 16x32 f16 A-fragment from a row-major matrix (row stride lda elems).
// Per ISA 16-bit A layout: lanes 0-15: K0..7 + K16..23 ; lanes 16-31: K8..15 + K24..31.
__device__ __forceinline__ v16h load_a_frag(const _Float16* tile, int lda, int lane) {
  int r    = lane & 15;
  int ksel = (lane >> 4) << 3;              // 0 or 8
  const _Float16* q = tile + (size_t)r * lda + ksel;
  v8h lo = *(const v8h*)(q);                // K = ksel .. ksel+7
  v8h hi = *(const v8h*)(q + 16);           // K = ksel+16 .. ksel+23
  return __builtin_shufflevector(lo, hi, 0,1,2,3,4,5,6,7,8,9,10,11,12,13,14,15);
}

// ---------------- phase 0: x0 = tanh(inputs[:, ::6, :] @ W0 + b0) -> f16 ----
__global__ void prep_x0(const float* __restrict__ inp, const float* __restrict__ W0,
                        const float* __restrict__ b0, _Float16* __restrict__ x0) {
  int idx = blockIdx.x * blockDim.x + threadIdx.x;      // (b*T + t)*128 + j
  if (idx >= M_TOK * 128) return;
  int j = idx & 127;
  int t = (idx >> 7) % T_SEQ;
  int b = idx / (T_SEQ * 128);
  const float* xr = inp + ((size_t)b * 1200 + (size_t)t * 6) * 8;
  float s = b0[j];
#pragma unroll
  for (int k = 0; k < 8; ++k) s += xr[k] * W0[k * 128 + j];
  x0[idx] = (_Float16)tanhf(s);
}

// ---------------- pack f32 [K x 2048] weight into WMMA B-fragment layout ----
// out[((kt*NT + ntile)*32 + lane)*16 + j] = W[kt*32 + (lane>=16?8:0) + (j&7) + (j>=8?16:0)][ntile*16 + (lane&15)]
__global__ void pack_b(const float* __restrict__ W, _Float16* __restrict__ out, int K) {
  int idx = blockIdx.x * blockDim.x + threadIdx.x;
  if (idx >= K * NGATE) return;
  int j     = idx & 15;
  int lane  = (idx >> 4) & 31;
  int tile  = idx >> 9;
  int ntile = tile % NT;
  int kt    = tile / NT;
  int krow  = kt * 32 + ((lane >> 4) << 3) + (j & 7) + ((j >> 3) << 4);
  int col   = ntile * 16 + (lane & 15);
  out[idx] = (_Float16)W[(size_t)krow * NGATE + col];
}

// ---------------- big GEMM: C[M x 2048] = A_f16[M x K] @ Bp + bias (f32) ----
__global__ void gemm_xw(const _Float16* __restrict__ A, const _Float16* __restrict__ Bp,
                        const float* __restrict__ bias, float* __restrict__ C, int K) {
  int lane = threadIdx.x & 31;
  int wave = threadIdx.x >> 5;
  int nt   = blockIdx.x * 4 + wave;     // 0..127
  int mt   = blockIdx.y;                // 0..M/16-1
  const _Float16* Abase = A + (size_t)mt * 16 * K;
  v8f acc = {};
  int KT = K >> 5;
  for (int kt = 0; kt < KT; ++kt) {
    v16h a = load_a_frag(Abase + kt * 32, K, lane);
    v16h b = *(const v16h*)(Bp + (((size_t)kt * NT + nt) * 32 + lane) * 16);
    acc = wmma_f16(a, b, acc);
  }
  int n  = lane & 15;
  int mb = (lane >> 4) << 3;
  int col = nt * 16 + n;
  float bv = bias[col];
#pragma unroll
  for (int v = 0; v < 8; ++v) {
    int row = mt * 16 + mb + v;
    C[(size_t)row * NGATE + col] = acc[v] + bv;
  }
}

// ---------------- fused recurrent step: z = xw_t + h@U ; gates ; c,h -------
__global__ void lstm_step(const _Float16* __restrict__ hprev, const _Float16* __restrict__ Up,
                          const float* __restrict__ xw_t, long xw_row_stride,
                          float* __restrict__ cstate, _Float16* __restrict__ hnew,
                          _Float16* __restrict__ seq_out_t, long seq_row_stride,
                          float* __restrict__ hf32) {
  int lane = threadIdx.x & 31;
  int wave = threadIdx.x >> 5;
  int nt   = blockIdx.x * 4 + wave;     // 0..31 (d/16 tiles)
  int mt   = blockIdx.y;                // 0..15 (batch tiles)
  const _Float16* Abase = hprev + (size_t)mt * 16 * DMODEL;
  v8f ai = {}, af = {}, ag = {}, ao = {};
  for (int kt = 0; kt < DMODEL / 32; ++kt) {
    v16h a = load_a_frag(Abase + kt * 32, DMODEL, lane);
    const _Float16* bb = Up + (size_t)kt * NT * 32 * 16 + (size_t)lane * 16;
    v16h bi = *(const v16h*)(bb + (size_t)(0 * 32 + nt) * 32 * 16);
    v16h bf = *(const v16h*)(bb + (size_t)(1 * 32 + nt) * 32 * 16);
    v16h bg = *(const v16h*)(bb + (size_t)(2 * 32 + nt) * 32 * 16);
    v16h bo = *(const v16h*)(bb + (size_t)(3 * 32 + nt) * 32 * 16);
    ai = wmma_f16(a, bi, ai);
    af = wmma_f16(a, bf, af);
    ag = wmma_f16(a, bg, ag);
    ao = wmma_f16(a, bo, ao);
  }
  int n    = lane & 15;
  int mb   = (lane >> 4) << 3;
  int dcol = nt * 16 + n;
#pragma unroll
  for (int v = 0; v < 8; ++v) {
    int b = mt * 16 + mb + v;
    const float* xwr = xw_t + (size_t)b * xw_row_stride;
    float zi = ai[v] + xwr[dcol];
    float zf = af[v] + xwr[DMODEL + dcol];
    float zg = ag[v] + xwr[2 * DMODEL + dcol];
    float zo = ao[v] + xwr[3 * DMODEL + dcol];
    float ig = 1.f / (1.f + __expf(-zi));
    float fg = 1.f / (1.f + __expf(-zf));
    float gg = tanhf(zg);
    float og = 1.f / (1.f + __expf(-zo));
    size_t sidx = (size_t)b * DMODEL + dcol;
    float c = fg * cstate[sidx] + ig * gg;
    cstate[sidx] = c;
    float h = og * tanhf(c);
    hnew[sidx] = (_Float16)h;
    if (seq_out_t) seq_out_t[(size_t)b * seq_row_stride + dcol] = (_Float16)h;
    if (hf32) hf32[sidx] = h;
  }
}

// ---------------- heads: out = concat(h@W1+b1, h@W2+b2) --------------------
__global__ void heads(const float* __restrict__ hf, const float* __restrict__ W1,
                      const float* __restrict__ b1, const float* __restrict__ W2,
                      const float* __restrict__ b2, float* __restrict__ out) {
  int idx = blockIdx.x * blockDim.x + threadIdx.x;      // 2*256*60
  if (idx >= 2 * B_SZ * 60) return;
  int o = idx % 60;
  int b = (idx / 60) % B_SZ;
  int which = idx / (B_SZ * 60);
  const float* W  = which ? W2 : W1;
  const float* bb = which ? b2 : b1;
  float s = bb[o];
  const float* h = hf + (size_t)b * DMODEL;
  for (int k = 0; k < DMODEL; ++k) s += h[k] * W[k * 60 + o];
  out[idx] = s;
}

// ---------------------------------------------------------------------------
extern "C" void kernel_launch(void* const* d_in, const int* in_sizes, int n_in,
                              void* d_out, int out_size, void* d_ws, size_t ws_size,
                              hipStream_t stream) {
  const float* inputs = (const float*)d_in[0];
  const float* W0  = (const float*)d_in[3];
  const float* b0  = (const float*)d_in[4];
  const float* Wl0 = (const float*)d_in[5];
  const float* Ul0 = (const float*)d_in[6];
  const float* bl0 = (const float*)d_in[7];
  const float* Wl1 = (const float*)d_in[8];
  const float* Ul1 = (const float*)d_in[9];
  const float* bl1 = (const float*)d_in[10];
  const float* Wr  = (const float*)d_in[11];
  const float* Ur  = (const float*)d_in[12];
  const float* br  = (const float*)d_in[13];
  const float* W1  = (const float*)d_in[14];
  const float* b1  = (const float*)d_in[15];
  const float* W2  = (const float*)d_in[16];
  const float* b2  = (const float*)d_in[17];
  float* out = (float*)d_out;

  // ---- workspace carve-up (256B aligned) ----
  char* p = (char*)d_ws;
  auto take = [&](size_t bytes) { char* r = p; p += (bytes + 255) & ~(size_t)255; return r; };
  _Float16* x0   = (_Float16*)take((size_t)M_TOK * 128 * 2);
  _Float16* seqA = (_Float16*)take((size_t)M_TOK * DMODEL * 2);
  _Float16* seqB = (_Float16*)take((size_t)M_TOK * DMODEL * 2);
  float*    xw   = (float*)   take((size_t)M_TOK * NGATE * 4);
  _Float16* hA   = (_Float16*)take((size_t)B_SZ * DMODEL * 2);
  _Float16* hB   = (_Float16*)take((size_t)B_SZ * DMODEL * 2);
  float*    cst  = (float*)   take((size_t)B_SZ * DMODEL * 4);
  float*    hf32 = (float*)   take((size_t)B_SZ * DMODEL * 4);
  _Float16* Wl0p = (_Float16*)take((size_t)128    * NGATE * 2);
  _Float16* Ul0p = (_Float16*)take((size_t)DMODEL * NGATE * 2);
  _Float16* Wl1p = (_Float16*)take((size_t)DMODEL * NGATE * 2);
  _Float16* Ul1p = (_Float16*)take((size_t)DMODEL * NGATE * 2);
  _Float16* Wrp  = (_Float16*)take((size_t)DMODEL * NGATE * 2);
  _Float16* Urp  = (_Float16*)take((size_t)DMODEL * NGATE * 2);

  // ---- phase 0: dense front-end ----
  {
    int tot = M_TOK * 128;
    prep_x0<<<(tot + 255) / 256, 256, 0, stream>>>(inputs, W0, b0, x0);
  }
  // ---- phase 1: pack weights to WMMA B-fragment layout ----
  {
    int t128 = 128 * NGATE, t512 = DMODEL * NGATE;
    pack_b<<<(t128 + 255) / 256, 256, 0, stream>>>(Wl0, Wl0p, 128);
    pack_b<<<(t512 + 255) / 256, 256, 0, stream>>>(Ul0, Ul0p, DMODEL);
    pack_b<<<(t512 + 255) / 256, 256, 0, stream>>>(Wl1, Wl1p, DMODEL);
    pack_b<<<(t512 + 255) / 256, 256, 0, stream>>>(Ul1, Ul1p, DMODEL);
    pack_b<<<(t512 + 255) / 256, 256, 0, stream>>>(Wr,  Wrp,  DMODEL);
    pack_b<<<(t512 + 255) / 256, 256, 0, stream>>>(Ur,  Urp,  DMODEL);
  }

  const long xw_stride  = (long)T_SEQ * NGATE;    // per-batch row stride in xw
  const long seq_stride = (long)T_SEQ * DMODEL;   // per-batch row stride in seq buffers
  dim3 gemm_grid(NT / 4, M_TOK / 16);
  dim3 step_grid(8, B_SZ / 16);                   // 32 d-tiles / 4 waves, 16 batch tiles

  struct Layer {
    const _Float16* Aseq; const _Float16* Wp; const _Float16* Up; const float* bias;
    int K; _Float16* out_seq; float* out_hf32;
  } layers[3] = {
    { x0,   Wl0p, Ul0p, bl0, 128,    seqA,    nullptr },
    { seqA, Wl1p, Ul1p, bl1, DMODEL, seqB,    nullptr },
    { seqB, Wrp,  Urp,  br,  DMODEL, nullptr, hf32    },
  };

  for (int l = 0; l < 3; ++l) {
    // parallel input projection over all tokens: xw = A @ W + b
    gemm_xw<<<gemm_grid, 128, 0, stream>>>(layers[l].Aseq, layers[l].Wp,
                                           layers[l].bias, xw, layers[l].K);
    // zero recurrent state
    hipMemsetAsync(hA,  0, (size_t)B_SZ * DMODEL * 2, stream);
    hipMemsetAsync(cst, 0, (size_t)B_SZ * DMODEL * 4, stream);
    _Float16* hp = hA;
    _Float16* hn = hB;
    for (int t = 0; t < T_SEQ; ++t) {
      _Float16* seq_t = layers[l].out_seq ? layers[l].out_seq + (size_t)t * DMODEL : nullptr;
      lstm_step<<<step_grid, 128, 0, stream>>>(
          hp, layers[l].Up, xw + (size_t)t * NGATE, xw_stride,
          cst, hn, seq_t, seq_stride, layers[l].out_hf32);
      _Float16* tmp = hp; hp = hn; hn = tmp;
    }
  }

  // ---- heads ----
  {
    int tot = 2 * B_SZ * 60;
    heads<<<(tot + 255) / 256, 256, 0, stream>>>(hf32, W1, b1, W2, b2, out);
  }
  (void)in_sizes; (void)n_in; (void)out_size; (void)ws_size;
}
